// GCRN_28595892257608
// MI455X (gfx1250) — matrix operand order repaired and checked by reading, not compiled, not run on hardware
//
#include <hip/hip_runtime.h>

// ---------------------------------------------------------------------------
// GCRN forward for MI455X (gfx1250, wave32, WMMA).
// T=12, B=4, N=4096, D=32, BN=B*N=16384.
// ---------------------------------------------------------------------------

#define TT    12
#define DD    32
#define BNROW 16384          // B*N
#define NN    4096
#define NB    4              // batches

typedef __attribute__((ext_vector_type(16))) __bf16 v16bf;
typedef __attribute__((ext_vector_type(8)))  float  v8f;

union Frag {
    v16bf        v;
    unsigned int w[8];
    int4         q[2];
};

// round-to-nearest-even f32 -> bf16 (bit trick; NaN edge ignored)
__device__ __forceinline__ unsigned short f2bf(float x) {
    unsigned int u = __float_as_uint(x);
    return (unsigned short)((u + 0x7FFFu + ((u >> 16) & 1u)) >> 16);
}
__device__ __forceinline__ unsigned int pk_bf16(float lo, float hi) {
    return (unsigned int)f2bf(lo) | ((unsigned int)f2bf(hi) << 16);
}
__device__ __forceinline__ float sigm(float x) {
    return __builtin_amdgcn_rcpf(1.0f + __expf(-x));
}
__device__ __forceinline__ float tanh_(float x) {
    return 2.0f * sigm(2.0f * x) - 1.0f;
}

__device__ __forceinline__ v8f wmma_bf16(const Frag& a, const Frag& b, v8f c) {
    return __builtin_amdgcn_wmma_f32_16x16x32_bf16(
        /*neg_a=*/false, a.v, /*neg_b=*/false, b.v,
        /*c_mod=*/(short)0, c, /*reuse_a=*/false, /*reuse_b=*/false);
}

// ---------------------------------------------------------------------------
// Kernel 1: LSTM over T steps + temporal-mean residual + BN partial stats.
// One wave handles 16 rows (M=16). Gate GEMMs (K=32) done with
// v_wmma_f32_16x16x32_bf16: gates = bias + x@Wih^T + h@Whh^T, 16 WMMA/step.
// Grid: 128 blocks x 256 threads (8 waves) -> 16384 rows.
// ---------------------------------------------------------------------------
__global__ void __launch_bounds__(256)
k_lstm(const float* __restrict__ spat,   // [T, BNROW, DD]
       const float* __restrict__ Wih,    // [128, 32]
       const float* __restrict__ Whh,    // [128, 32]
       const float* __restrict__ bih,    // [128]
       const float* __restrict__ bhh,    // [128]
       float* __restrict__ tempOut,      // [BNROW, DD]
       float* __restrict__ gsum,         // [32]
       float* __restrict__ gsq)          // [32]
{
    __shared__ float stageF[8][512];     // per-wave 16x32 f32 staging (bf16 aliased)
    __shared__ float chsum[32], chsq[32];

    const int wave = threadIdx.x >> 5;
    const int lane = threadIdx.x & 31;
    const int m    = lane & 15;
    const int half = lane >> 4;
    const int rowBase = blockIdx.x * 128 + wave * 16;
    const int row  = rowBase + m;
    const int ko   = 8 * half;

    if (threadIdx.x < 32) { chsum[threadIdx.x] = 0.0f; chsq[threadIdx.x] = 0.0f; }
    __syncthreads();

    // ---- pre-pack weights into bf16 B-fragments (B: lane = gate column) ----
    Frag  wihF[8], whhF[8];
    float biasv[8];
    const int kb = 8 * half;
#pragma unroll
    for (int g = 0; g < 8; ++g) {
        const int wrow = g * 16 + m;
        const float4* p = (const float4*)(Wih + wrow * 32 + kb);
        float4 a = p[0], b = p[1], c = p[4], d = p[5];
        wihF[g].w[0] = pk_bf16(a.x, a.y); wihF[g].w[1] = pk_bf16(a.z, a.w);
        wihF[g].w[2] = pk_bf16(b.x, b.y); wihF[g].w[3] = pk_bf16(b.z, b.w);
        wihF[g].w[4] = pk_bf16(c.x, c.y); wihF[g].w[5] = pk_bf16(c.z, c.w);
        wihF[g].w[6] = pk_bf16(d.x, d.y); wihF[g].w[7] = pk_bf16(d.z, d.w);
        p = (const float4*)(Whh + wrow * 32 + kb);
        a = p[0]; b = p[1]; c = p[4]; d = p[5];
        whhF[g].w[0] = pk_bf16(a.x, a.y); whhF[g].w[1] = pk_bf16(a.z, a.w);
        whhF[g].w[2] = pk_bf16(b.x, b.y); whhF[g].w[3] = pk_bf16(b.z, b.w);
        whhF[g].w[4] = pk_bf16(c.x, c.y); whhF[g].w[5] = pk_bf16(c.z, c.w);
        whhF[g].w[6] = pk_bf16(d.x, d.y); whhF[g].w[7] = pk_bf16(d.z, d.w);
        biasv[g] = bih[wrow] + bhh[wrow];
    }

    v8f cst0, cst1;
#pragma unroll
    for (int e = 0; e < 8; ++e) { cst0[e] = 0.0f; cst1[e] = 0.0f; }
    Frag hA;
#pragma unroll
    for (int e = 0; e < 8; ++e) hA.w[e] = 0u;

    float4 mac0 = {0,0,0,0}, mac1 = {0,0,0,0}, mac2 = {0,0,0,0}, mac3 = {0,0,0,0};
    float hv0[8], hv1[8];
    unsigned short* sbf = (unsigned short*)&stageF[wave][0];

    for (int t = 0; t < TT; ++t) {
        // -------- load x_t tile, accumulate temporal mean, pack A-fragment --
        const float4* xp = (const float4*)(spat + ((size_t)t * BNROW + row) * DD);
        float4 x0 = xp[2 * half], x1 = xp[2 * half + 1];
        float4 x2 = xp[2 * half + 4], x3 = xp[2 * half + 5];
        mac0.x += x0.x; mac0.y += x0.y; mac0.z += x0.z; mac0.w += x0.w;
        mac1.x += x1.x; mac1.y += x1.y; mac1.z += x1.z; mac1.w += x1.w;
        mac2.x += x2.x; mac2.y += x2.y; mac2.z += x2.z; mac2.w += x2.w;
        mac3.x += x3.x; mac3.y += x3.y; mac3.z += x3.z; mac3.w += x3.w;
        Frag xA;
        xA.w[0] = pk_bf16(x0.x, x0.y); xA.w[1] = pk_bf16(x0.z, x0.w);
        xA.w[2] = pk_bf16(x1.x, x1.y); xA.w[3] = pk_bf16(x1.z, x1.w);
        xA.w[4] = pk_bf16(x2.x, x2.y); xA.w[5] = pk_bf16(x2.z, x2.w);
        xA.w[6] = pk_bf16(x3.x, x3.y); xA.w[7] = pk_bf16(x3.z, x3.w);

        // -------- gates = bias + x@Wih^T + h@Whh^T  (16 WMMA) ---------------
        v8f acc[8];
#pragma unroll
        for (int g = 0; g < 8; ++g) {
            v8f a;
#pragma unroll
            for (int e = 0; e < 8; ++e) a[e] = biasv[g];
            a = wmma_bf16(xA, wihF[g], a);
            a = wmma_bf16(hA, whhF[g], a);
            acc[g] = a;
        }

        // -------- elementwise LSTM cell (i,f,g,o tiles align with channels) -
#pragma unroll
        for (int e = 0; e < 8; ++e) {
            float i0 = acc[0][e], f0 = acc[2][e], g0 = acc[4][e], o0 = acc[6][e];
            float c0 = sigm(f0) * cst0[e] + sigm(i0) * tanh_(g0);
            cst0[e] = c0;
            hv0[e] = sigm(o0) * tanh_(c0);
            float i1 = acc[1][e], f1 = acc[3][e], g1 = acc[5][e], o1 = acc[7][e];
            float c1 = sigm(f1) * cst1[e] + sigm(i1) * tanh_(g1);
            cst1[e] = c1;
            hv1[e] = sigm(o1) * tanh_(c1);
        }

        // -------- C-layout -> A-layout transpose of h via LDS (bf16) --------
        if (t < TT - 1) {
#pragma unroll
            for (int e = 0; e < 8; ++e) {
                const int rm = e + 8 * half;
                sbf[rm * 32 + m]      = f2bf(hv0[e]);
                sbf[rm * 32 + 16 + m] = f2bf(hv1[e]);
            }
            asm volatile("s_wait_dscnt 0" ::: "memory");
            const int4* hp = (const int4*)(sbf + m * 32);
            hA.q[0] = hp[half];
            hA.q[1] = hp[half + 2];
            asm volatile("s_wait_dscnt 0" ::: "memory");
        }
    }

    // -------- temp = h_final + mean_t(spat); store + BN partial stats -------
#pragma unroll
    for (int e = 0; e < 8; ++e) {
        const int rm = e + 8 * half;
        stageF[wave][rm * 32 + m]      = hv0[e];
        stageF[wave][rm * 32 + 16 + m] = hv1[e];
    }
    asm volatile("s_wait_dscnt 0" ::: "memory");
    const float4* fp = (const float4*)&stageF[wave][m * 32];
    float4 h0 = fp[2 * half], h1 = fp[2 * half + 1];
    float4 h2 = fp[2 * half + 4], h3 = fp[2 * half + 5];
    const float inv = 1.0f / 12.0f;
    float4 t0, t1, t2, t3;
    t0.x = h0.x + mac0.x * inv; t0.y = h0.y + mac0.y * inv;
    t0.z = h0.z + mac0.z * inv; t0.w = h0.w + mac0.w * inv;
    t1.x = h1.x + mac1.x * inv; t1.y = h1.y + mac1.y * inv;
    t1.z = h1.z + mac1.z * inv; t1.w = h1.w + mac1.w * inv;
    t2.x = h2.x + mac2.x * inv; t2.y = h2.y + mac2.y * inv;
    t2.z = h2.z + mac2.z * inv; t2.w = h2.w + mac2.w * inv;
    t3.x = h3.x + mac3.x * inv; t3.y = h3.y + mac3.y * inv;
    t3.z = h3.z + mac3.z * inv; t3.w = h3.w + mac3.w * inv;

    float* op = tempOut + (size_t)row * DD;
    *(float4*)(op + ko)      = t0;
    *(float4*)(op + ko + 4)  = t1;
    *(float4*)(op + ko + 16) = t2;
    *(float4*)(op + ko + 20) = t3;

    atomicAdd(&chsum[ko + 0], t0.x);  atomicAdd(&chsq[ko + 0], t0.x * t0.x);
    atomicAdd(&chsum[ko + 1], t0.y);  atomicAdd(&chsq[ko + 1], t0.y * t0.y);
    atomicAdd(&chsum[ko + 2], t0.z);  atomicAdd(&chsq[ko + 2], t0.z * t0.z);
    atomicAdd(&chsum[ko + 3], t0.w);  atomicAdd(&chsq[ko + 3], t0.w * t0.w);
    atomicAdd(&chsum[ko + 4], t1.x);  atomicAdd(&chsq[ko + 4], t1.x * t1.x);
    atomicAdd(&chsum[ko + 5], t1.y);  atomicAdd(&chsq[ko + 5], t1.y * t1.y);
    atomicAdd(&chsum[ko + 6], t1.z);  atomicAdd(&chsq[ko + 6], t1.z * t1.z);
    atomicAdd(&chsum[ko + 7], t1.w);  atomicAdd(&chsq[ko + 7], t1.w * t1.w);
    atomicAdd(&chsum[ko + 16], t2.x); atomicAdd(&chsq[ko + 16], t2.x * t2.x);
    atomicAdd(&chsum[ko + 17], t2.y); atomicAdd(&chsq[ko + 17], t2.y * t2.y);
    atomicAdd(&chsum[ko + 18], t2.z); atomicAdd(&chsq[ko + 18], t2.z * t2.z);
    atomicAdd(&chsum[ko + 19], t2.w); atomicAdd(&chsq[ko + 19], t2.w * t2.w);
    atomicAdd(&chsum[ko + 20], t3.x); atomicAdd(&chsq[ko + 20], t3.x * t3.x);
    atomicAdd(&chsum[ko + 21], t3.y); atomicAdd(&chsq[ko + 21], t3.y * t3.y);
    atomicAdd(&chsum[ko + 22], t3.z); atomicAdd(&chsq[ko + 22], t3.z * t3.z);
    atomicAdd(&chsum[ko + 23], t3.w); atomicAdd(&chsq[ko + 23], t3.w * t3.w);

    __syncthreads();
    if (threadIdx.x < 32) {
        atomicAdd(&gsum[threadIdx.x], chsum[threadIdx.x]);
        atomicAdd(&gsq[threadIdx.x],  chsq[threadIdx.x]);
    }
}

// ---------------------------------------------------------------------------
// Kernel 2: fold BN stats into per-channel scale/shift (both branches).
// ---------------------------------------------------------------------------
__global__ void k_bnfin(const float* __restrict__ statD, const float* __restrict__ statG,
                        const float* __restrict__ gammaD, const float* __restrict__ betaD,
                        const float* __restrict__ gammaG, const float* __restrict__ betaG,
                        float* __restrict__ bnD, float* __restrict__ bnG)
{
    const int t = threadIdx.x;
    const float invN = 1.0f / (float)BNROW;
    if (t < 32) {
        float mu  = statD[t] * invN;
        float var = statD[32 + t] * invN - mu * mu;
        float sc  = gammaD[t] * rsqrtf(var + 1e-5f);
        bnD[t] = sc; bnD[32 + t] = betaD[t] - mu * sc;
    } else {
        int c = t - 32;
        float mu  = statG[c] * invN;
        float var = statG[32 + c] * invN - mu * mu;
        float sc  = gammaG[c] * rsqrtf(var + 1e-5f);
        bnG[c] = sc; bnG[32 + c] = betaG[c] - mu * sc;
    }
}

// ---------------------------------------------------------------------------
// Kernel 3: res_D[row] = BN(temp_D[row]) . w_d + b_d
// ---------------------------------------------------------------------------
__global__ void __launch_bounds__(256)
k_resD(const float* __restrict__ tempD, const float* __restrict__ bnD,
       const float* __restrict__ w_d, const float* __restrict__ b_d,
       float* __restrict__ out)
{
    const int row = blockIdx.x * blockDim.x + threadIdx.x;
    const float* tp = tempD + (size_t)row * DD;
    float acc = 0.0f;
#pragma unroll
    for (int k = 0; k < DD; ++k) {
        float v = tp[k] * bnD[k] + bnD[32 + k];
        acc += v * w_d[k];
    }
    out[row] = acc + b_d[0];
}

// ---------------------------------------------------------------------------
// Kernel 4: BN(temp_G) and lin = BN(temp_G) @ w_g^T + b_g; both emitted as
// row-major bf16 (ready for WMMA A/B fragment loads). Wave per row, lane=chan.
// ---------------------------------------------------------------------------
__global__ void __launch_bounds__(256)
k_lin(const float* __restrict__ tempG, const float* __restrict__ bnG,
      const float* __restrict__ w_g, const float* __restrict__ b_g,
      unsigned short* __restrict__ tgnBf, unsigned short* __restrict__ linBf)
{
    const int wave = threadIdx.x >> 5;
    const int lane = threadIdx.x & 31;
    const int row  = blockIdx.x * 8 + wave;
    const float tgn = tempG[(size_t)row * DD + lane] * bnG[lane] + bnG[32 + lane];
    tgnBf[(size_t)row * DD + lane] = f2bf(tgn);
    float acc = b_g[lane];
#pragma unroll
    for (int k = 0; k < DD; ++k) {
        float tk = __shfl(tgn, k, 32);
        acc += w_g[lane * DD + k] * tk;
    }
    linBf[(size_t)row * DD + lane] = f2bf(acc);
}

// ---------------------------------------------------------------------------
// Kernel 5: res_G[b,i,j] = sum_d lin[b,i,d] * tempGn[b,j,d]
// Wave computes a 16x128 strip: 1 A-frag, 8 B-frags, 8 WMMA (K=32 = full D).
// Grid: (32 j-tiles, 32 i-tiles, 4 batches) x 256 threads. Store-bound.
// ---------------------------------------------------------------------------
__global__ void __launch_bounds__(256)
k_einsum(const unsigned short* __restrict__ linBf,
         const unsigned short* __restrict__ tgnBf,
         float* __restrict__ outG)
{
    const int wave = threadIdx.x >> 5;
    const int lane = threadIdx.x & 31;
    const int m    = lane & 15;
    const int half = lane >> 4;
    const int b     = blockIdx.z;
    const int iBase = blockIdx.y * 128 + wave * 16;
    const int jBase = blockIdx.x * 128;

    const unsigned short* lb = linBf + (size_t)b * NN * DD;
    const unsigned short* tb = tgnBf + (size_t)b * NN * DD;

    Frag A;
    {
        const int4* ap = (const int4*)(lb + (size_t)(iBase + m) * DD);
        A.q[0] = ap[half];       // K = 8*half .. 8*half+7
        A.q[1] = ap[half + 2];   // K = 8*half+16 .. +23
    }

    float* orow = outG + ((size_t)b * NN + iBase) * NN + jBase;
#pragma unroll
    for (int jt = 0; jt < 8; ++jt) {
        Frag Bf;
        const int4* bp = (const int4*)(tb + (size_t)(jBase + jt * 16 + m) * DD);
        Bf.q[0] = bp[half];      // K = 8*half .. 8*half+7
        Bf.q[1] = bp[half + 2];  // K = 8*half+16 .. +23
        v8f c;
#pragma unroll
        for (int e = 0; e < 8; ++e) c[e] = 0.0f;
        c = wmma_bf16(A, Bf, c);
        const int j = jt * 16 + m;
#pragma unroll
        for (int r = 0; r < 8; ++r) {
            const int irow = r + 8 * half;
            orow[(size_t)irow * NN + j] = c[r];
        }
    }
}

// ---------------------------------------------------------------------------
extern "C" void kernel_launch(void* const* d_in, const int* in_sizes, int n_in,
                              void* d_out, int out_size, void* d_ws, size_t ws_size,
                              hipStream_t stream)
{
    const float* spatD  = (const float*)d_in[0];
    const float* spatG  = (const float*)d_in[1];
    const float* WihD   = (const float*)d_in[2];
    const float* WhhD   = (const float*)d_in[3];
    const float* bihD   = (const float*)d_in[4];
    const float* bhhD   = (const float*)d_in[5];
    const float* WihG   = (const float*)d_in[6];
    const float* WhhG   = (const float*)d_in[7];
    const float* bihG   = (const float*)d_in[8];
    const float* bhhG   = (const float*)d_in[9];
    const float* gammaD = (const float*)d_in[10];
    const float* betaD  = (const float*)d_in[11];
    const float* gammaG = (const float*)d_in[12];
    const float* betaG  = (const float*)d_in[13];
    const float* w_d    = (const float*)d_in[14];
    const float* b_d    = (const float*)d_in[15];
    const float* w_g    = (const float*)d_in[16];
    const float* b_g    = (const float*)d_in[17];

    float* ws    = (float*)d_ws;
    float* tempD = ws;                      // 524288 f32
    float* tempG = ws + 524288;             // 524288 f32
    float* statD = ws + 1048576;            // 64 f32 (sum[32], sumsq[32])
    float* statG = statD + 64;              // 64 f32
    float* bnD   = statG + 64;              // 64 f32 (scale[32], shift[32])
    float* bnG   = bnD + 64;                // 64 f32
    unsigned short* tgnBf = (unsigned short*)(bnG + 64);   // 524288 bf16
    unsigned short* linBf = tgnBf + (size_t)BNROW * DD;    // 524288 bf16

    float* outD = (float*)d_out;            // [B,N] = 16384
    float* outG = outD + BNROW;             // [B,N,N] = 67108864

    hipMemsetAsync(statD, 0, 128 * sizeof(float), stream);

    k_lstm<<<128, 256, 0, stream>>>(spatD, WihD, WhhD, bihD, bhhD,
                                    tempD, statD, statD + 32);
    k_lstm<<<128, 256, 0, stream>>>(spatG, WihG, WhhG, bihG, bhhG,
                                    tempG, statG, statG + 32);
    k_bnfin<<<1, 64, 0, stream>>>(statD, statG, gammaD, betaD, gammaG, betaG,
                                  bnD, bnG);
    k_resD<<<64, 256, 0, stream>>>(tempD, bnD, w_d, b_d, outD);
    k_lin<<<2048, 256, 0, stream>>>(tempG, bnG, w_g, b_g, tgnBf, linBf);
    k_einsum<<<dim3(32, 32, NB), 256, 0, stream>>>(linBf, tgnBf, outG);
}
